// HymbaAttention_89378269429884
// MI455X (gfx1250) — matrix operand order repaired and checked by reading, not compiled, not run on hardware
//
#include <hip/hip_runtime.h>

// ---------------------------------------------------------------------------
// Problem constants (match reference)
// ---------------------------------------------------------------------------
constexpr int BATCH = 2;
constexpr int SEQ   = 2048;
constexpr int DIMC  = 1024;
constexpr int NH    = 16;     // query heads
constexpr int NKVH  = 4;      // kv heads
constexpr int HD    = 64;     // head dim
constexpr int KVD   = NKVH * HD;   // 256
constexpr int ROPE  = 32;
constexpr int NST   = 8;      // SSM state dim
constexpr int DTR   = 64;     // dt rank
constexpr int MROWS = BATCH * SEQ; // 4096

// ---------------------------------------------------------------------------
// WMMA types / helpers (gfx1250, wave32)
// ---------------------------------------------------------------------------
typedef __bf16  v16bf __attribute__((ext_vector_type(16)));
typedef float   v8f   __attribute__((ext_vector_type(8)));
typedef unsigned int u32x8 __attribute__((ext_vector_type(8)));

__device__ __forceinline__ v16bf frag16(const __bf16* p0, const __bf16* p1) {
  // Two contiguous 16B chunks -> one 32B bf16x16 fragment.
  uint4 a = *(const uint4*)p0;
  uint4 b = *(const uint4*)p1;
  u32x8 r;
  r[0]=a.x; r[1]=a.y; r[2]=a.z; r[3]=a.w;
  r[4]=b.x; r[5]=b.y; r[6]=b.z; r[7]=b.w;
  return __builtin_bit_cast(v16bf, r);
}

__device__ __forceinline__ v16bf frag16m(const __bf16* p0, const __bf16* p1, unsigned m) {
  uint4 a = *(const uint4*)p0;
  uint4 b = *(const uint4*)p1;
  u32x8 r;
  r[0]=a.x&m; r[1]=a.y&m; r[2]=a.z&m; r[3]=a.w&m;
  r[4]=b.x&m; r[5]=b.y&m; r[6]=b.z&m; r[7]=b.w&m;
  return __builtin_bit_cast(v16bf, r);
}

__device__ __forceinline__ v8f zero8() {
  v8f z;
#pragma unroll
  for (int i = 0; i < 8; ++i) z[i] = 0.0f;
  return z;
}

__device__ __forceinline__ float sigmoidf_(float x) { return 1.0f / (1.0f + __expf(-x)); }

// ---------------------------------------------------------------------------
// fp32 -> bf16 conversion
// ---------------------------------------------------------------------------
__global__ void cvt_bf16_kernel(const float* __restrict__ in, __bf16* __restrict__ out, int n) {
  int i = blockIdx.x * blockDim.x + threadIdx.x;
  if (i < n) out[i] = (__bf16)in[i];
}

// ---------------------------------------------------------------------------
// Generic GEMM:  C[M,Nc](f32) = A[M,K](bf16, row-major) * W[Nc,K](bf16)^T
// Wave tile 32x64: 2 A-frags share 4 B-frags -> 8 WMMAs per 32-wide k-step
// (~21 flop/byte of operand traffic per lane).  Block = 8 waves = 256 rows.
// FULL=true: Nc multiple of 64, no column masking in the hot loop.
// Fragment layouts per CDNA5 ISA 7.12.2.
// ---------------------------------------------------------------------------
template<bool FULL>
__global__ __launch_bounds__(256)
void gemm_bf16_wmma(const __bf16* __restrict__ A, const __bf16* __restrict__ W,
                    float* __restrict__ C, int Nc, int K, int ldc) {
  const int lane = threadIdx.x & 31;
  const int wid  = threadIdx.x >> 5;
  const int l16  = lane & 15;
  const int half = lane >> 4;
  const int row0 = blockIdx.x * 256 + wid * 32;
  const int col0 = blockIdx.y * 64;

  // A: lanes 0-31 map M = lane&15; elems 0..7 = K (half*8 .. +7), 8..15 = +16
  const __bf16* arow0 = A + (size_t)(row0 + l16) * K;
  const __bf16* arow1 = A + (size_t)(row0 + 16 + l16) * K;

  // B: lane holds column n = col0 + nf*16 + l16; elems 0..7 = K half*8..+7, 8..15 = +16
  const __bf16* wp[4];
  unsigned wm[4];
#pragma unroll
  for (int nf = 0; nf < 4; ++nf) {
    int n = col0 + nf * 16 + l16;
    bool v = FULL || (n < Nc);
    wp[nf] = W + (size_t)(v ? n : 0) * K;
    wm[nf] = v ? 0xFFFFFFFFu : 0u;
  }

  v8f acc0[4], acc1[4];
#pragma unroll
  for (int nf = 0; nf < 4; ++nf) { acc0[nf] = zero8(); acc1[nf] = zero8(); }

  for (int kk = 0; kk < K; kk += 32) {
    v16bf af0 = frag16(arow0 + kk + half * 8, arow0 + kk + 16 + half * 8);
    v16bf af1 = frag16(arow1 + kk + half * 8, arow1 + kk + 16 + half * 8);
    __builtin_prefetch(arow0 + kk + 256, 0, 3);   // global_prefetch_b8 (near)
    __builtin_prefetch(arow1 + kk + 256, 0, 3);
#pragma unroll
    for (int nf = 0; nf < 4; ++nf) {
      const __bf16* bp = wp[nf] + kk + half * 8;
      v16bf bf_ = FULL ? frag16(bp, bp + 16) : frag16m(bp, bp + 16, wm[nf]);
      acc0[nf] = __builtin_amdgcn_wmma_f32_16x16x32_bf16(
          false, af0, false, bf_, (short)0, acc0[nf], false, false);
      acc1[nf] = __builtin_amdgcn_wmma_f32_16x16x32_bf16(
          false, af1, false, bf_, (short)0, acc1[nf], false, false);
    }
  }

  // C layout: VGPR i, lanes 0-15 -> M=i, lanes 16-31 -> M=8+i; N = l16
#pragma unroll
  for (int nf = 0; nf < 4; ++nf) {
    int c = col0 + nf * 16 + l16;
    if (FULL || c < Nc) {
#pragma unroll
      for (int i = 0; i < 8; ++i) {
        int r = row0 + half * 8 + i;
        C[(size_t)r * ldc + c] = acc0[nf][i];
        C[(size_t)(r + 16) * ldc + c] = acc1[nf][i];
      }
    }
  }
}

// ---------------------------------------------------------------------------
// Per-head RMS norm + partial RoPE (+optional gain), fp32 in -> bf16 out.
// One wave per (row, head); lane handles dims {lane, lane+32}.
// ---------------------------------------------------------------------------
__global__ __launch_bounds__(256)
void rms_rope_kernel(const float* __restrict__ X, const float* __restrict__ gain,
                     __bf16* __restrict__ Y, int nheads) {
  const int lane = threadIdx.x & 31;
  const int wid  = threadIdx.x >> 5;
  const int gw   = blockIdx.x * 8 + wid;
  const int row  = gw / nheads;
  const int h    = gw % nheads;
  const int pos  = row % SEQ;

  const float* x = X + (size_t)row * nheads * HD + h * HD;
  float x0 = x[lane];
  float x1 = x[lane + 32];
  float ss = x0 * x0 + x1 * x1;
#pragma unroll
  for (int m = 1; m < 32; m <<= 1) ss += __shfl_xor(ss, m, 32);
  float rms = rsqrtf(ss * (1.0f / 64.0f) + 1.1920929e-07f);
  float n0 = x0 * rms, n1 = x1 * rms;

  // RoPE on dims [0,32): pairs (fi, fi+16), fi = lane & 15.
  int fi = lane & 15;
  float inv = __expf(-((float)(2 * fi) / (float)ROPE) * 9.210340371976184f); // ln(1e4)
  float ang = (float)pos * inv;
  float c = __cosf(ang), s = __sinf(ang);
  float partner = __shfl_xor(n0, 16, 32);
  float r0 = (lane < 16) ? (n0 * c + partner * s) : (n0 * c - partner * s);

  float g = gain ? gain[h] : 1.0f;
  __bf16* y = Y + (size_t)row * nheads * HD + h * HD;
  y[lane]      = (__bf16)(r0 * g);
  y[lane + 32] = (__bf16)(n1 * g);
}

// v = v_proj * sigmoid(gate_proj), fp32 -> bf16
__global__ void vgate_kernel(const float* __restrict__ v, const float* __restrict__ g,
                             __bf16* __restrict__ out, int n) {
  int i = blockIdx.x * blockDim.x + threadIdx.x;
  if (i < n) out[i] = (__bf16)(v[i] * sigmoidf_(g[i]));
}

// ---------------------------------------------------------------------------
// Flash-style causal GQA attention.  One wave owns a 16-query tile of one
// head and streams keys in chunks of 32.  QK^T and P*V both use
// v_wmma_f32_16x16x32_bf16; P is moved C-layout -> A-layout through LDS,
// V tiles staged in LDS for the strided B-fragment gather.
// ---------------------------------------------------------------------------
__global__ __launch_bounds__(128)
void attn_kernel(const __bf16* __restrict__ Q, const __bf16* __restrict__ Kb,
                 const __bf16* __restrict__ Vb, float* __restrict__ O) {
  __shared__ __align__(16) __bf16 s_p[4][16 * 32];
  __shared__ __align__(16) __bf16 s_v[4][32 * 64];

  const int lane = threadIdx.x & 31;
  const int wid  = threadIdx.x >> 5;
  const int l16  = lane & 15;
  const int half = lane >> 4;
  const int S16  = SEQ / 16;

  int gw = blockIdx.x * 4 + wid;
  int b  = gw / (NH * S16);
  int r  = gw % (NH * S16);
  int h  = r / S16;
  int q0 = (r % S16) * 16;
  int kvh = h / (NH / NKVH);

  __bf16* pb = s_p[wid];
  __bf16* vt = s_v[wid];

  // Q A-fragments (M = lane&15 row, dims split per ISA A-layout)
  const __bf16* qrow = Q + ((size_t)(b * SEQ + q0 + l16) * DIMC + h * HD);
  v16bf aq[2];
#pragma unroll
  for (int kc = 0; kc < 2; ++kc)
    aq[kc] = frag16(qrow + kc * 32 + half * 8, qrow + kc * 32 + 16 + half * 8);

  v8f oacc[4];
#pragma unroll
  for (int nf = 0; nf < 4; ++nf) oacc[nf] = zero8();
  float mrun[8], lrun[8];
#pragma unroll
  for (int i = 0; i < 8; ++i) { mrun[i] = -3.0e38f; lrun[i] = 0.0f; }

  for (int k0 = 0; k0 < q0 + 16; k0 += 32) {
    // ---- scores: two 16x16 tiles (32 keys) -------------------------------
    float sc2[2][8];
#pragma unroll
    for (int sf = 0; sf < 2; ++sf) {
      int key  = k0 + sf * 16 + l16;
      int keyc = key < SEQ ? key : SEQ - 1;
      const __bf16* kp = Kb + ((size_t)(b * SEQ + keyc) * KVD + kvh * HD);
      v8f sa = zero8();
#pragma unroll
      for (int kc = 0; kc < 2; ++kc) {
        const __bf16* p = kp + kc * 32 + half * 8;
        v16bf bk = frag16(p, p + 16);
        sa = __builtin_amdgcn_wmma_f32_16x16x32_bf16(
            false, aq[kc], false, bk, (short)0, sa, false, false);
      }
#pragma unroll
      for (int i = 0; i < 8; ++i) {
        float sv = sa[i] * 0.125f;               // 1/sqrt(64)
        int qi = q0 + half * 8 + i;
        if (key > qi) sv = -1.0e30f;             // causal mask
        sc2[sf][i] = sv;
      }
    }

    // ---- stage V tile (32 keys x dims 0..31) in LDS -------------------------
    {
      int kv  = k0 + lane;
      int kvc = kv < SEQ ? kv : SEQ - 1;
      const __bf16* vr = Vb + ((size_t)(b * SEQ + kvc) * KVD + kvh * HD);
#pragma unroll
      for (int j = 0; j < 4; ++j)
        *(uint4*)(vt + lane * 64 + j * 8) = *(const uint4*)(vr + j * 8);
    }

    // ---- online softmax (rows live in 16-lane half-groups) ---------------
#pragma unroll
    for (int i = 0; i < 8; ++i) {
      float t = fmaxf(sc2[0][i], sc2[1][i]);
#pragma unroll
      for (int m = 1; m < 16; m <<= 1) t = fmaxf(t, __shfl_xor(t, m, 32));
      float mn = fmaxf(mrun[i], t);
      float p0 = __expf(sc2[0][i] - mn);
      float p1 = __expf(sc2[1][i] - mn);
      float ps = p0 + p1;
#pragma unroll
      for (int m = 1; m < 16; m <<= 1) ps += __shfl_xor(ps, m, 32);
      float scl = __expf(mrun[i] - mn);
      lrun[i] = lrun[i] * scl + ps;
      mrun[i] = mn;
#pragma unroll
      for (int nf = 0; nf < 4; ++nf) oacc[nf][i] *= scl;
      int rr = half * 8 + i;
      pb[rr * 32 + l16]      = (__bf16)p0;
      pb[rr * 32 + 16 + l16] = (__bf16)p1;
    }
    asm volatile("s_wait_dscnt 0" ::: "memory");

    // ---- P A-fragment from LDS ------------------------------------------
    const __bf16* pr = pb + l16 * 32;
    v16bf pa = frag16(pr + half * 16, pr + half * 16 + 8);

    // ---- P(16x32) @ V(32x64) --------------------------------------------
#pragma unroll
    for (int nf = 0; nf < 4; ++nf) {
      int d = nf * 16 + l16;
      u32x8 bu;
#pragma unroll
      for (int j = 0; j < 8; ++j) {
        unsigned lo = *(const unsigned short*)(vt + (half * 16 + 2 * j)     * 64 + d);
        unsigned hi = *(const unsigned short*)(vt + (half * 16 + 2 * j + 1) * 64 + d);
        bu[j] = lo | (hi << 16);
      }
      v16bf bv = __builtin_bit_cast(v16bf, bu);
      oacc[nf] = __builtin_amdgcn_wmma_f32_16x16x32_bf16(
          false, pa, false, bv, (short)0, oacc[nf], false, false);
    }
    asm volatile("s_wait_dscnt 0" ::: "memory");  // WAR on LDS before next iter
  }

  // ---- epilogue: normalize and store fp32 --------------------------------
#pragma unroll
  for (int nf = 0; nf < 4; ++nf) {
    int col = h * HD + nf * 16 + l16;
#pragma unroll
    for (int i = 0; i < 8; ++i) {
      int row = q0 + half * 8 + i;
      O[((size_t)b * SEQ + row) * DIMC + col] = oacc[nf][i] / lrun[i];
    }
  }
}

// ---------------------------------------------------------------------------
// Depthwise causal conv(3) + SiLU.  Input x_ssm lives in cols [0,1024) of the
// (M,2048) him-slice buffer.  Outputs fp32 (for scan) and bf16 (for hxp GEMM).
// ---------------------------------------------------------------------------
__global__ void conv_silu_kernel(const float* __restrict__ sg, const float* __restrict__ cw,
                                 const float* __restrict__ cb, float* __restrict__ uf,
                                 __bf16* __restrict__ ub) {
  int i = blockIdx.x * blockDim.x + threadIdx.x;
  if (i >= BATCH * SEQ * DIMC) return;
  int c = i % DIMC;
  int s = (i / DIMC) % SEQ;
  int b = i / (DIMC * SEQ);
  const float* xc = sg + (size_t)b * SEQ * 2048 + c;   // stride 2048 per step
  float acc = cb[c];
  float w0 = cw[c * 3], w1 = cw[c * 3 + 1], w2 = cw[c * 3 + 2];
  if (s >= 2) acc += w0 * xc[(size_t)(s - 2) * 2048];
  if (s >= 1) acc += w1 * xc[(size_t)(s - 1) * 2048];
  acc += w2 * xc[(size_t)s * 2048];
  float u = acc * sigmoidf_(acc);
  uf[i] = u;
  ub[i] = (__bf16)u;
}

// ---------------------------------------------------------------------------
// Selective scan: one thread per (b, channel); sequential over SEQ with
// N=8 states.  B/C (shared across channels of a batch) staged in LDS.
// Fused: + u*D, * silu(gate), -> bf16 for the hmo GEMM.
// ---------------------------------------------------------------------------
__global__ __launch_bounds__(256)
void ssm_scan_kernel(const float* __restrict__ uf, const float* __restrict__ sp,
                     const float* __restrict__ sg, const float* __restrict__ A_log,
                     const float* __restrict__ Dp, const float* __restrict__ dtb,
                     __bf16* __restrict__ yg) {
  __shared__ float sB[NST], sC[NST];
  const int b = blockIdx.x / (DIMC / 256);
  const int c = (blockIdx.x % (DIMC / 256)) * 256 + threadIdx.x;

  float a[NST], hn[NST];
#pragma unroll
  for (int n = 0; n < NST; ++n) {
    a[n] = -__expf(A_log[c * NST + n]);
    hn[n] = 0.0f;
  }
  const float dpar = Dp[c];
  const float bias = dtb[c];

  for (int s = 0; s < SEQ; ++s) {
    size_t pbase = ((size_t)b * SEQ + s) * 80;
    if (threadIdx.x < NST)            sB[threadIdx.x]       = sp[pbase + DTR + threadIdx.x];
    else if (threadIdx.x < 2 * NST)   sC[threadIdx.x - NST] = sp[pbase + DTR + NST + (threadIdx.x - NST)];
    __syncthreads();

    float dtv = (c < DTR) ? sp[pbase + c] : 0.0f;
    float xv = dtv + bias;
    float delta = (xv > 20.0f) ? xv : log1pf(__expf(xv));   // softplus
    float uv = uf[((size_t)b * SEQ + s) * DIMC + c];
    float du = delta * uv;
    float y = 0.0f;
#pragma unroll
    for (int n = 0; n < NST; ++n) {
      hn[n] = __expf(delta * a[n]) * hn[n] + du * sB[n];
      y += hn[n] * sC[n];
    }
    y += uv * dpar;
    float gv = sg[((size_t)b * SEQ + s) * 2048 + 1024 + c]; // gate columns
    y *= gv * sigmoidf_(gv);
    yg[((size_t)b * SEQ + s) * DIMC + c] = (__bf16)y;
    __syncthreads();
  }
}

// merged = bf16( ya*w + ym*(1-w) ), w = sigmoid(merge_alpha)
__global__ void merge_kernel(const float* __restrict__ ya, const float* __restrict__ ym,
                             const float* __restrict__ alpha, __bf16* __restrict__ out, int n) {
  int i = blockIdx.x * blockDim.x + threadIdx.x;
  if (i < n) {
    float w = sigmoidf_(alpha[0]);
    out[i] = (__bf16)(ya[i] * w + ym[i] * (1.0f - w));
  }
}

// ---------------------------------------------------------------------------
// Host orchestration
// ---------------------------------------------------------------------------
static inline int ceil_div(int a, int b) { return (a + b - 1) / b; }

extern "C" void kernel_launch(void* const* d_in, const int* in_sizes, int n_in,
                              void* d_out, int out_size, void* d_ws, size_t ws_size,
                              hipStream_t stream) {
  (void)in_sizes; (void)n_in; (void)out_size; (void)ws_size;

  const float* x_f      = (const float*)d_in[0];
  const float* q_w      = (const float*)d_in[1];
  const float* k_w      = (const float*)d_in[2];
  const float* v_w      = (const float*)d_in[3];
  const float* out_w    = (const float*)d_in[4];
  const float* him_w    = (const float*)d_in[5];
  const float* hvg_w    = (const float*)d_in[6];
  const float* hxp_w    = (const float*)d_in[7];
  const float* hmo_w    = (const float*)d_in[8];
  const float* q_gain   = (const float*)d_in[9];
  const float* conv_w   = (const float*)d_in[10];
  const float* conv_b   = (const float*)d_in[11];
  const float* A_log    = (const float*)d_in[12];
  const float* D_param  = (const float*)d_in[13];
  const float* dt_bias  = (const float*)d_in[14];
  const float* m_alpha  = (const float*)d_in[15];
  float* outp = (float*)d_out;

  // ---- bump allocator over d_ws (~173 MB total) -----------------------------
  size_t off = 0;
  auto alloc = [&](size_t bytes) -> char* {
    off = (off + 255) & ~(size_t)255;
    char* p = (char*)d_ws + off;
    off += bytes;
    return p;
  };
  const size_t M = MROWS;
  __bf16* xb    = (__bf16*)alloc(M * DIMC * 2);
  __bf16* wq    = (__bf16*)alloc((size_t)DIMC * DIMC * 2);
  __bf16* wk    = (__bf16*)alloc((size_t)KVD * DIMC * 2);
  __bf16* wv    = (__bf16*)alloc((size_t)KVD * DIMC * 2);
  __bf16* wg    = (__bf16*)alloc((size_t)KVD * DIMC * 2);
  __bf16* whim  = (__bf16*)alloc((size_t)2048 * DIMC * 2);
  __bf16* whxp  = (__bf16*)alloc((size_t)80 * DIMC * 2);
  __bf16* whmo  = (__bf16*)alloc((size_t)DIMC * DIMC * 2);
  __bf16* wout  = (__bf16*)alloc((size_t)DIMC * DIMC * 2);
  float*  qf    = (float*)alloc(M * DIMC * 4);
  float*  kf    = (float*)alloc(M * KVD * 4);
  float*  vf    = (float*)alloc(M * KVD * 4);
  float*  gf    = (float*)alloc(M * KVD * 4);
  __bf16* qb    = (__bf16*)alloc(M * DIMC * 2);
  __bf16* kb    = (__bf16*)alloc(M * KVD * 2);
  __bf16* vb    = (__bf16*)alloc(M * KVD * 2);
  float*  sgb   = (float*)alloc(M * 2048 * 4);     // x_ssm | gate
  float*  uf    = (float*)alloc(M * DIMC * 4);
  __bf16* ub    = (__bf16*)alloc(M * DIMC * 2);
  float*  ssmp  = (float*)alloc(M * 80 * 4);
  float*  ya    = (float*)alloc(M * DIMC * 4);
  __bf16* yg    = (__bf16*)alloc(M * DIMC * 2);
  float*  ym    = (float*)alloc(M * DIMC * 4);
  __bf16* mg    = (__bf16*)alloc(M * DIMC * 2);

  auto cvt = [&](const float* src, __bf16* dst, int n) {
    cvt_bf16_kernel<<<ceil_div(n, 256), 256, 0, stream>>>(src, dst, n);
  };

  // ---- convert activations + weights to bf16 --------------------------------
  cvt(x_f, xb, M * DIMC);
  cvt(q_w, wq, DIMC * DIMC);
  cvt(k_w, wk, KVD * DIMC);
  cvt(v_w, wv, KVD * DIMC);
  cvt(hvg_w, wg, KVD * DIMC);
  cvt(him_w + (size_t)(2 * KVD) * DIMC, whim, 2048 * DIMC);  // rows 512..2559
  cvt(hxp_w, whxp, 80 * DIMC);
  cvt(hmo_w, whmo, DIMC * DIMC);
  cvt(out_w, wout, DIMC * DIMC);

  auto gemm = [&](const __bf16* A, const __bf16* W, float* C, int Nc) {
    dim3 grid(MROWS / 256, ceil_div(Nc, 64));
    if (Nc % 64 == 0)
      gemm_bf16_wmma<true><<<grid, 256, 0, stream>>>(A, W, C, Nc, DIMC, Nc);
    else
      gemm_bf16_wmma<false><<<grid, 256, 0, stream>>>(A, W, C, Nc, DIMC, Nc);
  };

  // ---- projections ----------------------------------------------------------
  gemm(xb, wq, qf, DIMC);
  gemm(xb, wk, kf, KVD);
  gemm(xb, wv, vf, KVD);
  gemm(xb, wg, gf, KVD);
  gemm(xb, whim, sgb, 2048);

  // ---- norm / rope / gates --------------------------------------------------
  rms_rope_kernel<<<(MROWS * NH) / 8, 256, 0, stream>>>(qf, q_gain, qb, NH);
  rms_rope_kernel<<<(MROWS * NKVH) / 8, 256, 0, stream>>>(kf, nullptr, kb, NKVH);
  vgate_kernel<<<ceil_div(MROWS * KVD, 256), 256, 0, stream>>>(vf, gf, vb, MROWS * KVD);

  // ---- attention ------------------------------------------------------------
  attn_kernel<<<(BATCH * NH * (SEQ / 16)) / 4, 128, 0, stream>>>(qb, kb, vb, ya);

  // ---- SSM branch -----------------------------------------------------------
  conv_silu_kernel<<<ceil_div(MROWS * DIMC, 256), 256, 0, stream>>>(sgb, conv_w, conv_b, uf, ub);
  gemm(ub, whxp, ssmp, 80);
  ssm_scan_kernel<<<BATCH * (DIMC / 256), 256, 0, stream>>>(uf, ssmp, sgb, A_log, D_param, dt_bias, yg);
  gemm(yg, whmo, ym, DIMC);

  // ---- merge + output projection -------------------------------------------
  merge_kernel<<<ceil_div(MROWS * DIMC, 256), 256, 0, stream>>>(ya, ym, m_alpha, mg, MROWS * DIMC);
  gemm(mg, wout, outp, DIMC);
}
